// GlobalAttentionDecoder_13383118094942
// MI455X (gfx1250) — matrix operand order repaired and checked by reading, not compiled, not run on hardware
//
#include <hip/hip_runtime.h>
#include <hip/hip_bf16.h>

typedef __attribute__((ext_vector_type(16))) __bf16 v16bf;
typedef __attribute__((ext_vector_type(8)))  __bf16 v8bf;
typedef __attribute__((ext_vector_type(8)))  float  v8f;

#define HH   1000
#define HP   1024
#define NB   64
#define LLEN 50
#define VV   50000
#define TT   52
#define NSTEP 51

__device__ __forceinline__ __bf16 f2bf(float x) { return (__bf16)x; }

__device__ __forceinline__ v16bf combine8(v8bf lo, v8bf hi) {
  v16bf r;
#pragma unroll
  for (int i = 0; i < 8; ++i) { r[i] = lo[i]; r[i + 8] = hi[i]; }
  return r;
}

// A fragment (16x32 bf16), row-major A with leading dim lda, rows m0..m0+15, k base kb.
// ISA 7.12.2: lane = m, VGPR0..3 hold K = hh*8 + 0..7, VGPR4..7 hold K = 16 + hh*8 + 0..7.
__device__ __forceinline__ v16bf load_a_frag(const __bf16* A, int lda, int m0, int kb, int lane) {
  int m  = m0 + (lane & 15);
  int hh = lane >> 4;
  const __bf16* p = A + (size_t)m * lda + kb + hh * 8;
  return combine8(*(const v8bf*)p, *(const v8bf*)(p + 16));
}

// B fragment (32x16 bf16) sourced from row-major weight W[n,k] (n = output col), ldw = K stride.
// lane = n, elements 0..7 hold K = kb + hh*8 + 0..7, elements 8..15 hold K = kb + 16 + hh*8 + 0..7.
__device__ __forceinline__ v16bf load_b_frag(const __bf16* W, int ldw, int n0, int kb, int lane) {
  int n  = n0 + (lane & 15);
  int hh = lane >> 4;
  const __bf16* p = W + (size_t)n * ldw + kb + hh * 8;
  return combine8(*(const v8bf*)p, *(const v8bf*)(p + 16));
}

// ---------------------------------------------------------------------------
// Generic WMMA GEMM: C[M x Nout] = A[M x Kp] * W[Nout x Kp]^T (+bias)
// block = 128 threads (4 waves), workgroup tile = 64 rows x 128 cols,
// each wave owns 32 cols (2 B frags, 8 accumulators) so every A fragment
// feeds 2 WMMAs. mode 1: tanh epilogue + bf16 secondary output.
// ---------------------------------------------------------------------------
__global__ void wmma_gemm(const __bf16* __restrict__ A, const __bf16* __restrict__ W,
                          const float* __restrict__ bias, float* __restrict__ C,
                          __bf16* __restrict__ Cb, int Nout, int Kp,
                          long long ldc, int ldcb, int mode) {
  int lane = threadIdx.x & 31;
  int wid  = threadIdx.x >> 5;
  int mbase = blockIdx.x * 64;
  int nA   = blockIdx.y * 128 + wid * 32;     // first 16-col subtile
  if (nA >= Nout) return;
  bool has2 = (nA + 16 < Nout);
  int nB   = has2 ? (nA + 16) : nA;           // clamped duplicate if tail

  v8f acc0[4], acc1[4];
#pragma unroll
  for (int t = 0; t < 4; ++t)
#pragma unroll
    for (int r = 0; r < 8; ++r) { acc0[t][r] = 0.f; acc1[t][r] = 0.f; }

  for (int kb = 0; kb < Kp; kb += 32) {
    v16bf b0 = load_b_frag(W, Kp, nA, kb, lane);
    v16bf b1 = load_b_frag(W, Kp, nB, kb, lane);
#pragma unroll
    for (int t = 0; t < 4; ++t) {
      v16bf af = load_a_frag(A, Kp, mbase + t * 16, kb, lane);
      acc0[t] = __builtin_amdgcn_wmma_f32_16x16x32_bf16(false, af, false, b0,
                                                        (short)0, acc0[t], false, false);
      acc1[t] = __builtin_amdgcn_wmma_f32_16x16x32_bf16(false, af, false, b1,
                                                        (short)0, acc1[t], false, false);
    }
  }

  int nc0 = nA + (lane & 15);
  float bv0 = bias ? bias[nc0] : 0.f;
#pragma unroll
  for (int t = 0; t < 4; ++t) {
#pragma unroll
    for (int r = 0; r < 8; ++r) {
      int m   = mbase + t * 16 + (lane >> 4) * 8 + r;
      float v = acc0[t][r] + bv0;
      if (mode == 1) v = tanhf(v);
      C[(size_t)m * ldc + nc0] = v;
      if (Cb) Cb[(size_t)m * ldcb + nc0] = f2bf(v);
    }
  }
  if (has2) {
    int nc1 = nB + (lane & 15);
    float bv1 = bias ? bias[nc1] : 0.f;
#pragma unroll
    for (int t = 0; t < 4; ++t) {
#pragma unroll
      for (int r = 0; r < 8; ++r) {
        int m   = mbase + t * 16 + (lane >> 4) * 8 + r;
        float v = acc1[t][r] + bv1;
        if (mode == 1) v = tanhf(v);
        C[(size_t)m * ldc + nc1] = v;
        if (Cb) Cb[(size_t)m * ldcb + nc1] = f2bf(v);
      }
    }
  }
}

// ---------------------------------------------------------------------------
// Fused LSTM step: gates = X*Wih^T + Hprev*Whh^T + b, then cell elementwise.
// block = 128 (4 waves: one gate quadrant each, 32 cols), grid = HP/32 = 32.
// ---------------------------------------------------------------------------
__global__ void lstm_step(const __bf16* __restrict__ X, int Kx,
                          const __bf16* __restrict__ Hprev,
                          const __bf16* __restrict__ Wih, const __bf16* __restrict__ Whh,
                          const float* __restrict__ bias,
                          float* __restrict__ h_out, float* __restrict__ c_st,
                          __bf16* __restrict__ hb_out) {
  __shared__ float g[4][64][32];
  int lane = threadIdx.x & 31;
  int q    = threadIdx.x >> 5;           // gate quadrant i,f,g,o
  int j0   = blockIdx.x * 32;
  int n0   = q * HP + j0;                // first subtile row inside padded weight
  int n1   = n0 + 16;

  v8f acc0[4], acc1[4];
#pragma unroll
  for (int t = 0; t < 4; ++t)
#pragma unroll
    for (int r = 0; r < 8; ++r) { acc0[t][r] = 0.f; acc1[t][r] = 0.f; }

  for (int kb = 0; kb < Kx; kb += 32) {
    v16bf b0 = load_b_frag(Wih, Kx, n0, kb, lane);
    v16bf b1 = load_b_frag(Wih, Kx, n1, kb, lane);
#pragma unroll
    for (int t = 0; t < 4; ++t) {
      v16bf af = load_a_frag(X, Kx, t * 16, kb, lane);
      acc0[t] = __builtin_amdgcn_wmma_f32_16x16x32_bf16(false, af, false, b0,
                                                        (short)0, acc0[t], false, false);
      acc1[t] = __builtin_amdgcn_wmma_f32_16x16x32_bf16(false, af, false, b1,
                                                        (short)0, acc1[t], false, false);
    }
  }
  for (int kb = 0; kb < HP; kb += 32) {
    v16bf b0 = load_b_frag(Whh, HP, n0, kb, lane);
    v16bf b1 = load_b_frag(Whh, HP, n1, kb, lane);
#pragma unroll
    for (int t = 0; t < 4; ++t) {
      v16bf af = load_a_frag(Hprev, HP, t * 16, kb, lane);
      acc0[t] = __builtin_amdgcn_wmma_f32_16x16x32_bf16(false, af, false, b0,
                                                        (short)0, acc0[t], false, false);
      acc1[t] = __builtin_amdgcn_wmma_f32_16x16x32_bf16(false, af, false, b1,
                                                        (short)0, acc1[t], false, false);
    }
  }

  float bv0 = bias[n0 + (lane & 15)];
  float bv1 = bias[n1 + (lane & 15)];
#pragma unroll
  for (int t = 0; t < 4; ++t)
#pragma unroll
    for (int r = 0; r < 8; ++r) {
      int m = t * 16 + (lane >> 4) * 8 + r;
      g[q][m][(lane & 15)]      = acc0[t][r] + bv0;
      g[q][m][(lane & 15) + 16] = acc1[t][r] + bv1;
    }
  __syncthreads();

#pragma unroll
  for (int e = 0; e < 16; ++e) {
    int id  = threadIdx.x + 128 * e;     // 0..2047
    int m   = id >> 5;
    int jj  = id & 31;
    int col = blockIdx.x * 32 + jj;
    float gi = g[0][m][jj], gf = g[1][m][jj], gg = g[2][m][jj], go = g[3][m][jj];
    float si = 1.f / (1.f + expf(-gi));
    float sf = 1.f / (1.f + expf(-gf));
    float so = 1.f / (1.f + expf(-go));
    float cp = c_st[m * HP + col];
    float cn = sf * cp + si * tanhf(gg);
    float hn = so * tanhf(cn);
    c_st[m * HP + col]   = cn;
    h_out[m * HP + col]  = hn;
    hb_out[m * HP + col] = f2bf(hn);
  }
}

// ---------------------------------------------------------------------------
// Elementwise / prep kernels
// ---------------------------------------------------------------------------
__global__ void conv_pad(const float* __restrict__ src, __bf16* __restrict__ dst,
                         int rows, int cols, int pcols, long long total) {
  long long idx = (long long)blockIdx.x * 256 + threadIdx.x;
  if (idx >= total) return;
  int r = (int)(idx / pcols), c = (int)(idx % pcols);
  float v = (r < rows && c < cols) ? src[(size_t)r * cols + c] : 0.f;
  dst[idx] = f2bf(v);
}

// LSTM weight remap: src rows laid out in 4 quadrants of 1000 -> padded quadrants of 1024
__global__ void conv_lstm_w(const float* __restrict__ src, __bf16* __restrict__ dst,
                            int cols, int pcols, long long total) {
  long long idx = (long long)blockIdx.x * 256 + threadIdx.x;
  if (idx >= total) return;
  int pr = (int)(idx / pcols), c = (int)(idx % pcols);
  int q = pr >> 10, r = pr & 1023;
  float v = (r < HH && c < cols) ? src[(size_t)(q * HH + r) * cols + c] : 0.f;
  dst[idx] = f2bf(v);
}

__global__ void comb_bias(const float* __restrict__ bih, const float* __restrict__ bhh,
                          float* __restrict__ out) {
  int pr = blockIdx.x * 256 + threadIdx.x;
  if (pr >= 4096) return;
  int q = pr >> 10, r = pr & 1023;
  out[pr] = (r < HH) ? (bih[q * HH + r] + bhh[q * HH + r]) : 0.f;
}

__global__ void init_state(const float* __restrict__ h0in, const float* __restrict__ c0in,
                           float* __restrict__ h0, float* __restrict__ c0,
                           float* __restrict__ h1, float* __restrict__ c1,
                           float* __restrict__ attn, __bf16* __restrict__ h0b,
                           __bf16* __restrict__ h1b) {
  int n = blockIdx.x;
  for (int c = threadIdx.x; c < HP; c += 256) {
    float vh0 = (c < HH) ? h0in[n * HH + c] : 0.f;
    float vh1 = (c < HH) ? h0in[NB * HH + n * HH + c] : 0.f;
    float vc0 = (c < HH) ? c0in[n * HH + c] : 0.f;
    float vc1 = (c < HH) ? c0in[NB * HH + n * HH + c] : 0.f;
    h0[n * HP + c] = vh0;  h1[n * HP + c] = vh1;
    c0[n * HP + c] = vc0;  c1[n * HP + c] = vc1;
    attn[n * HP + c] = 0.f;
    h0b[n * HP + c] = f2bf(vh0);
    h1b[n * HP + c] = f2bf(vh1);
  }
}

__global__ void embed_concat(const int* __restrict__ target, int t,
                             const float* __restrict__ emb, const float* __restrict__ attn,
                             __bf16* __restrict__ x0b) {
  int n   = blockIdx.x;
  int tok = target[n * TT + t];
  for (int p = threadIdx.x; p < 2048; p += 256) {
    float v;
    if (p < HH)          v = emb[(size_t)tok * HH + p];
    else if (p < 2 * HH) v = attn[n * HP + (p - HH)];
    else                 v = 0.f;
    x0b[n * 2048 + p] = f2bf(v);
  }
}

// scores -> softmax -> ctx -> cat = [ctx, h1] in bf16 (one block per batch row)
__global__ void attention_k(const float* __restrict__ h1, const float* __restrict__ encp,
                            const float* __restrict__ enc, __bf16* __restrict__ catb) {
  __shared__ float sc[64];
  __shared__ float pr[64];
  int n    = blockIdx.x;
  int lane = threadIdx.x & 31;
  int w    = threadIdx.x >> 5;   // wave id 0..7
  for (int l = w; l < LLEN; l += 8) {
    float s = 0.f;
    const float* hp = h1 + n * HP;
    const float* ep = encp + ((size_t)n * LLEN + l) * HP;
    for (int h = lane; h < HH; h += 32) s += hp[h] * ep[h];
#pragma unroll
    for (int off = 16; off; off >>= 1) s += __shfl_xor(s, off, 32);
    if (lane == 0) sc[l] = s;
  }
  __syncthreads();
  if (threadIdx.x == 0) {
    float mx = sc[0];
    for (int l = 1; l < LLEN; ++l) mx = fmaxf(mx, sc[l]);
    float sum = 0.f;
    for (int l = 0; l < LLEN; ++l) { float e = expf(sc[l] - mx); pr[l] = e; sum += e; }
    float inv = 1.f / sum;
    for (int l = 0; l < LLEN; ++l) pr[l] *= inv;
  }
  __syncthreads();
  for (int c = threadIdx.x; c < HH; c += 256) {
    float ctx = 0.f;
    const float* ep = enc + (size_t)n * LLEN * HH + c;
    for (int l = 0; l < LLEN; ++l) ctx += pr[l] * ep[(size_t)l * HH];
    catb[n * 2048 + c]      = f2bf(ctx);
    catb[n * 2048 + HH + c] = f2bf(h1[n * HP + c]);
  }
  for (int p = 2 * HH + threadIdx.x; p < 2048; p += 256) catb[n * 2048 + p] = f2bf(0.f);
}

__global__ void write_hc(const float* __restrict__ h0, const float* __restrict__ h1,
                         const float* __restrict__ c0, const float* __restrict__ c1,
                         float* __restrict__ out) {
  int idx   = blockIdx.x * 256 + threadIdx.x;
  int total = 2 * NB * HH;
  if (idx >= total) return;
  int s = idx / (NB * HH), rem = idx % (NB * HH);
  int n = rem / HH, c = rem % HH;
  out[idx]         = (s == 0 ? h0 : h1)[n * HP + c];
  out[total + idx] = (s == 0 ? c0 : c1)[n * HP + c];
}

// ---------------------------------------------------------------------------
extern "C" void kernel_launch(void* const* d_in, const int* in_sizes, int n_in,
                              void* d_out, int out_size, void* d_ws, size_t ws_size,
                              hipStream_t stream) {
  const float* enc    = (const float*)d_in[0];
  const float* h_0    = (const float*)d_in[1];
  const float* c_0    = (const float*)d_in[2];
  const int*   target = (const int*)d_in[3];
  const float* emb    = (const float*)d_in[4];
  const float* Wih0   = (const float*)d_in[5];
  const float* Whh0   = (const float*)d_in[6];
  const float* bih0   = (const float*)d_in[7];
  const float* bhh0   = (const float*)d_in[8];
  const float* Wih1   = (const float*)d_in[9];
  const float* Whh1   = (const float*)d_in[10];
  const float* bih1   = (const float*)d_in[11];
  const float* bhh1   = (const float*)d_in[12];
  const float* Wa     = (const float*)d_in[13];
  const float* Wc     = (const float*)d_in[14];
  const float* Wout   = (const float*)d_in[15];
  float* out = (float*)d_out;

  char* ws = (char*)d_ws;
  size_t off = 0;
  auto alloc = [&](size_t bytes) -> void* {
    void* p = ws + off;
    off += (bytes + 255) & ~(size_t)255;
    return p;
  };
  __bf16* Wih0b = (__bf16*)alloc((size_t)4096 * 2048 * 2);
  __bf16* Whh0b = (__bf16*)alloc((size_t)4096 * 1024 * 2);
  __bf16* Wih1b = (__bf16*)alloc((size_t)4096 * 1024 * 2);
  __bf16* Whh1b = (__bf16*)alloc((size_t)4096 * 1024 * 2);
  __bf16* Wcb   = (__bf16*)alloc((size_t)1024 * 2048 * 2);
  __bf16* Wab   = (__bf16*)alloc((size_t)1024 * 1024 * 2);
  __bf16* Woutb = (__bf16*)alloc((size_t)VV * 1024 * 2);
  __bf16* encb  = (__bf16*)alloc((size_t)NB * LLEN * 1024 * 2);
  float*  b0    = (float*)alloc(4096 * 4);
  float*  b1    = (float*)alloc(4096 * 4);
  float*  encp  = (float*)alloc((size_t)NB * LLEN * 1024 * 4);
  float*  h0s   = (float*)alloc((size_t)NB * HP * 4);
  float*  c0s   = (float*)alloc((size_t)NB * HP * 4);
  float*  h1s   = (float*)alloc((size_t)NB * HP * 4);
  float*  c1s   = (float*)alloc((size_t)NB * HP * 4);
  float*  attn  = (float*)alloc((size_t)NB * HP * 4);
  __bf16* x0b   = (__bf16*)alloc((size_t)NB * 2048 * 2);
  __bf16* catb  = (__bf16*)alloc((size_t)NB * 2048 * 2);
  __bf16* attnb = (__bf16*)alloc((size_t)NB * HP * 2);
  __bf16* h0b[2], *h1b[2];
  h0b[0] = (__bf16*)alloc((size_t)NB * HP * 2);
  h0b[1] = (__bf16*)alloc((size_t)NB * HP * 2);
  h1b[0] = (__bf16*)alloc((size_t)NB * HP * 2);
  h1b[1] = (__bf16*)alloc((size_t)NB * HP * 2);

  // ---- weight prep (bf16 + padding; everything ends up L2-resident ~150MB) ----
  long long tot;
  tot = 4096LL * 2048;
  conv_lstm_w<<<(unsigned)((tot + 255) / 256), 256, 0, stream>>>(Wih0, Wih0b, 2000, 2048, tot);
  tot = 4096LL * 1024;
  conv_lstm_w<<<(unsigned)((tot + 255) / 256), 256, 0, stream>>>(Whh0, Whh0b, 1000, 1024, tot);
  conv_lstm_w<<<(unsigned)((tot + 255) / 256), 256, 0, stream>>>(Wih1, Wih1b, 1000, 1024, tot);
  conv_lstm_w<<<(unsigned)((tot + 255) / 256), 256, 0, stream>>>(Whh1, Whh1b, 1000, 1024, tot);
  tot = 1024LL * 2048;
  conv_pad<<<(unsigned)((tot + 255) / 256), 256, 0, stream>>>(Wc, Wcb, 1000, 2000, 2048, tot);
  tot = 1024LL * 1024;
  conv_pad<<<(unsigned)((tot + 255) / 256), 256, 0, stream>>>(Wa, Wab, 1000, 1000, 1024, tot);
  tot = (long long)VV * 1024;
  conv_pad<<<(unsigned)((tot + 255) / 256), 256, 0, stream>>>(Wout, Woutb, VV, 1000, 1024, tot);
  tot = (long long)NB * LLEN * 1024;
  conv_pad<<<(unsigned)((tot + 255) / 256), 256, 0, stream>>>(enc, encb, NB * LLEN, 1000, 1024, tot);
  comb_bias<<<16, 256, 0, stream>>>(bih0, bhh0, b0);
  comb_bias<<<16, 256, 0, stream>>>(bih1, bhh1, b1);
  init_state<<<NB, 256, 0, stream>>>(h_0, c_0, h0s, c0s, h1s, c1s, attn, h0b[0], h1b[0]);

  // enc_proj = enc @ W_a^T : (3200 x 1024) via WMMA
  wmma_gemm<<<dim3(NB * LLEN / 64, 1024 / 128), 128, 0, stream>>>(
      encb, Wab, nullptr, encp, nullptr, 1024, 1024, 1024LL, 0, 0);

  // ---- 51 sequential decode steps ----
  for (int t = 0; t < NSTEP; ++t) {
    embed_concat<<<NB, 256, 0, stream>>>(target, t, emb, attn, x0b);
    lstm_step<<<HP / 32, 128, 0, stream>>>(x0b, 2048, h0b[t & 1], Wih0b, Whh0b, b0,
                                           h0s, c0s, h0b[(t + 1) & 1]);
    lstm_step<<<HP / 32, 128, 0, stream>>>(h0b[(t + 1) & 1], 1024, h1b[t & 1], Wih1b, Whh1b, b1,
                                           h1s, c1s, h1b[(t + 1) & 1]);
    attention_k<<<NB, 256, 0, stream>>>(h1s, encp, enc, catb);
    // attn_new = tanh(cat @ W_c^T)  (also emits bf16 copy for the logits GEMM)
    wmma_gemm<<<dim3(1, 1024 / 128), 128, 0, stream>>>(
        catb, Wcb, nullptr, attn, attnb, 1024, 2048, 1024LL, 1024, 1);
    // logits -> d_out[n, t, :]
    wmma_gemm<<<dim3(1, (VV + 127) / 128), 128, 0, stream>>>(
        attnb, Woutb, nullptr, out + (size_t)t * VV, nullptr, VV, 1024,
        (long long)NSTEP * VV, 0, 0);
  }

  // ---- h_T / c_T tail ----
  float* tail = out + (size_t)NB * NSTEP * VV;
  write_hc<<<(2 * NB * HH + 255) / 256, 256, 0, stream>>>(h0s, h1s, c0s, c1s, tail);
}
